// MultiHeadAttention_53309134078537
// MI455X (gfx1250) — matrix operand order repaired and checked by reading, not compiled, not run on hardware
//
#include <hip/hip_runtime.h>
#include <hip/hip_bf16.h>

// ---------------------------------------------------------------------------
// MHA forward for MI455X (gfx1250, wave32, WMMA 16x16x32 bf16, f32 accum,
// async global->LDS staging with per-lane transpose scatter)
// B=4, S=2048, D=512, H=8, HD=64
// ---------------------------------------------------------------------------

typedef __bf16 bf16_t;
typedef __attribute__((ext_vector_type(16))) __bf16 v16bf;
typedef __attribute__((ext_vector_type(8)))  __bf16 v8bf;
typedef __attribute__((ext_vector_type(8)))  float  v8f;

#define MHA_B  4
#define MHA_S  2048
#define MHA_D  512
#define MHA_H  8
#define MHA_HD 64

// padded LDS row strides (multiples of 8 floats -> keep 32B alignment for v8f)
#define ASTR 40     // 16-row f32 A tile stride
#define BSTR 40     // 128-row f32 B^T tile stride
#define PSTR 2056   // 16 x 2048 logits panel stride

static __device__ __forceinline__ v8f wmma_bf16(v16bf a, v16bf b, v8f c) {
    return __builtin_amdgcn_wmma_f32_16x16x32_bf16(false, a, false, b,
                                                   (short)0, c, false, false);
}

static __device__ __forceinline__ v16bf cat8(v8bf lo, v8bf hi) {
    return __builtin_shufflevector(lo, hi, 0, 1, 2, 3, 4, 5, 6, 7,
                                           8, 9, 10, 11, 12, 13, 14, 15);
}

// pack two v8f into one bf16 x16 fragment (elements [0..7]=x, [8..15]=y)
static __device__ __forceinline__ v16bf cvtf32x16(v8f x, v8f y) {
    v16bf r;
#pragma unroll
    for (int i = 0; i < 8; ++i) {
        r[i]     = (bf16_t)x[i];
        r[i + 8] = (bf16_t)y[i];
    }
    return r;
}

// issue one async b32 global->LDS copy (per-lane addresses; ASYNCcnt tracked)
static __device__ __forceinline__ void async_b32(const void* gptr, void* lptr) {
    unsigned long long ga = (unsigned long long)gptr;
    unsigned int       la = (unsigned int)(unsigned long long)lptr; // low 32 = LDS offset
    asm volatile("global_load_async_to_lds_b32 %0, %1, off"
                 :: "v"(la), "v"(ga) : "memory");
}

static __device__ __forceinline__ void async_wait0() {
    asm volatile("s_wait_asynccnt 0x0" ::: "memory");
}

// ---------------------------------------------------------------------------
// Kernel 1: P = X @ W + bias  (X f32 (8192,512), W f32 (512,512))
//   transposed == 0 : store bf16 at (B,H,S,HD)  [q, k]
//   transposed == 1 : store bf16 at (B,H,HD,S)  [v]
// grid = (512, 4), block = 256 (8 waves; wave w owns a 16-col strip)
// ---------------------------------------------------------------------------
__global__ __launch_bounds__(256)
void proj_kernel(const float* __restrict__ X, const float* __restrict__ W,
                 const float* __restrict__ bias, bf16_t* __restrict__ dst,
                 int transposed) {
    __shared__ float lAf[16 * ASTR];    // [r][k]   f32
    __shared__ float lBt[128 * BSTR];   // [n][k]   f32 (transposed by async scatter)

    const int tid  = threadIdx.x;
    const int w    = tid >> 5;
    const int lane = tid & 31;
    const int r    = lane & 15;
    const int hh   = lane >> 4;

    const int m0 = blockIdx.x * 16;
    const int n0 = blockIdx.y * 128;

    v8f acc = {};

    for (int k0 = 0; k0 < MHA_D; k0 += 32) {
        // --- async stage A tile 16x32 f32 (2 elements / thread) ------------
#pragma unroll
        for (int t = 0; t < 2; ++t) {
            int i  = tid + t * 256;
            int rr = i >> 5, kk = i & 31;
            async_b32(X + (size_t)(m0 + rr) * MHA_D + k0 + kk,
                      &lAf[rr * ASTR + kk]);
        }
        // --- async stage B tile 32x128 f32, transposed into [n][k] ---------
        // global reads coalesced along n; LDS scatter does the transpose.
#pragma unroll
        for (int t = 0; t < 16; ++t) {
            int i  = tid + t * 256;
            int nn = i & 127, kk = i >> 7;
            async_b32(W + (size_t)(k0 + kk) * MHA_D + n0 + nn,
                      &lBt[nn * BSTR + kk]);
        }
        async_wait0();
        __syncthreads();

        // A fragment: lane (r,hh): K = hh*16 .. +15
        const v8f* pa = (const v8f*)(lAf + r * ASTR);
        v16bf af = cvtf32x16(pa[2 * hh], pa[2 * hh + 1]);
        // B fragment: lane (n=r,hh): row w*16+r of B^T, k = hh*16 .. +15
        const v8f* pb = (const v8f*)(lBt + (w * 16 + r) * BSTR);
        v16bf bfr = cvtf32x16(pb[2 * hh], pb[2 * hh + 1]);

        acc = wmma_bf16(af, bfr, acc);
        __syncthreads();
    }

    // epilogue: C vgpr t, lane (r,hh) -> row t + hh*8, col r
#pragma unroll
    for (int t = 0; t < 8; ++t) {
        int row = t + hh * 8;
        int gm  = m0 + row;
        int gn  = n0 + w * 16 + r;
        float val = acc[t] + bias[gn];
        int b = gm >> 11, s = gm & (MHA_S - 1);
        int hx = gn >> 6, hd = gn & (MHA_HD - 1);
        size_t idx = transposed
            ? ((((size_t)b * MHA_H + hx) * MHA_HD + hd) * MHA_S + s)
            : ((((size_t)b * MHA_H + hx) * MHA_S + s) * MHA_HD + hd);
        dst[idx] = (bf16_t)val;
    }
}

// ---------------------------------------------------------------------------
// Kernel 2: attention core. grid=(S/16, H, B), block=256.
// Dynamic LDS: P[16*PSTR] f32 logits/probs, red[256], accbuf[8*8*32].
// ---------------------------------------------------------------------------
__global__ __launch_bounds__(256)
void attn_kernel(const bf16_t* __restrict__ qb, const bf16_t* __restrict__ kb,
                 const bf16_t* __restrict__ vt, const float* __restrict__ pad_mask,
                 float* __restrict__ attn_out, bf16_t* __restrict__ zb) {
    extern __shared__ float smem[];
    float* P      = smem;                    // 16 x PSTR
    float* red    = smem + 16 * PSTR;        // 16 x 16
    float* accbuf = red + 256;               // 8 waves x 8 x 32

    const int tid  = threadIdx.x;
    const int w    = tid >> 5;
    const int lane = tid & 31;
    const int r    = lane & 15;
    const int hh   = lane >> 4;

    const int q0 = blockIdx.x * 16;
    const int h  = blockIdx.y;
    const int b  = blockIdx.z;

    const bf16_t* qbase = qb + ((size_t)b * MHA_H + h) * MHA_S * MHA_HD; // [s][hd]
    const bf16_t* kbase = kb + ((size_t)b * MHA_H + h) * MHA_S * MHA_HD; // [s][hd]
    const bf16_t* vbase = vt + ((size_t)b * MHA_H + h) * MHA_HD * MHA_S; // [hd][s]

    // ---- Phase 1: logits = (q k^T) * scale + mask -------------------------
    const v8bf* qp = (const v8bf*)(qbase + (size_t)(q0 + r) * MHA_HD);
    v16bf aq0 = cat8(qp[2 * hh],     qp[2 * hh + 1]);   // hd 0..31
    v16bf aq1 = cat8(qp[4 + 2 * hh], qp[5 + 2 * hh]);   // hd 32..63

    const float scale = 0.125f;   // 1/sqrt(64)

    for (int tt = 0; tt < 16; ++tt) {
        int kcol0 = w * 256 + tt * 16;
        const v16bf* kp = (const v16bf*)(kbase + (size_t)(kcol0 + r) * MHA_HD);
        v16bf bk0 = kp[hh];       // hd hh*16 .. +15
        v16bf bk1 = kp[2 + hh];   // hd 32 + hh*16 .. +15
        v8f lacc = {};
        lacc = wmma_bf16(aq0, bk0, lacc);
        lacc = wmma_bf16(aq1, bk1, lacc);
        float mv = pad_mask[(size_t)b * MHA_S + kcol0 + r] * -1e9f;
#pragma unroll
        for (int t = 0; t < 8; ++t) {
            P[(t + hh * 8) * PSTR + kcol0 + r] = lacc[t] * scale + mv;
        }
    }
    __syncthreads();

    // ---- Phase 2: softmax over k, write attn f32 --------------------------
    {
        int row = tid >> 4, sub = tid & 15;
        float* Prow = P + row * PSTR;
        float m = -INFINITY;
        for (int c = sub; c < MHA_S; c += 16) m = fmaxf(m, Prow[c]);
        red[row * 16 + sub] = m;
        __syncthreads();
        m = red[row * 16];
#pragma unroll
        for (int i = 1; i < 16; ++i) m = fmaxf(m, red[row * 16 + i]);
        float ssum = 0.0f;
        for (int c = sub; c < MHA_S; c += 16) {
            float e = __expf(Prow[c] - m);
            Prow[c] = e;
            ssum += e;
        }
        __syncthreads();
        red[row * 16 + sub] = ssum;
        __syncthreads();
        ssum = 0.0f;
#pragma unroll
        for (int i = 0; i < 16; ++i) ssum += red[row * 16 + i];
        float inv = 1.0f / ssum;
        float* arow = attn_out + (((size_t)b * MHA_H + h) * MHA_S + q0 + row) * MHA_S;
        for (int c = sub; c < MHA_S; c += 16) {
            float p = Prow[c] * inv;
            Prow[c] = p;        // keep normalized probs in LDS for phase 3
            arow[c] = p;        // stream attention matrix to HBM
        }
    }
    __syncthreads();

    // ---- Phase 3: z = probs @ v  (8 waves = 4 hd-tiles x 2 K-halves) ------
    {
        int hd0   = (w & 3) * 16;
        int khalf = w >> 2;
        v8f zacc = {};
        for (int kk = 0; kk < 32; ++kk) {
            int kb2 = khalf * 1024 + kk * 32;
            if (kk + 1 < 32) {
                __builtin_prefetch(vbase + (size_t)(hd0 + r) * MHA_S + kb2 + 32, 0, 0);
            }
            // A fragment from LDS probs: two v8f loads + pack-convert
            const v8f* pr = (const v8f*)(P + r * PSTR + kb2);
            v16bf ap = cvtf32x16(pr[2 * hh], pr[2 * hh + 1]);
            // B fragment from v^T: lane n=r -> col hd0+n, 16 contiguous k
            const v16bf* vp = (const v16bf*)(vbase + (size_t)(hd0 + r) * MHA_S + kb2);
            v16bf bv = vp[hh];
            zacc = wmma_bf16(ap, bv, zacc);
        }
#pragma unroll
        for (int t = 0; t < 8; ++t) accbuf[(w * 8 + t) * 32 + lane] = zacc[t];
        __syncthreads();
        if (w < 4) {
#pragma unroll
            for (int t = 0; t < 8; ++t) {
                float vsum = accbuf[(w * 8 + t) * 32 + lane]
                           + accbuf[((w + 4) * 8 + t) * 32 + lane];
                int row = t + hh * 8;
                zb[((size_t)b * MHA_S + q0 + row) * MHA_D
                   + (size_t)h * MHA_HD + w * 16 + r] = (bf16_t)vsum;
            }
        }
    }
}

// ---------------------------------------------------------------------------
// Kernel 3: out = z @ Wout + bias   (z bf16 (8192,512), out f32 (8192,512))
// grid = (512, 4), block = 256
// ---------------------------------------------------------------------------
__global__ __launch_bounds__(256)
void outproj_kernel(const bf16_t* __restrict__ Z, const float* __restrict__ W,
                    const float* __restrict__ bias, float* __restrict__ out) {
    __shared__ bf16_t lA[16 * 32];      // [r][k] bf16 (raw async copy)
    __shared__ float  lBt[128 * BSTR];  // [n][k] f32 (transposed async scatter)

    const int tid  = threadIdx.x;
    const int w    = tid >> 5;
    const int lane = tid & 31;
    const int r    = lane & 15;
    const int hh   = lane >> 4;

    const int m0 = blockIdx.x * 16;
    const int n0 = blockIdx.y * 128;

    v8f acc = {};

    for (int k0 = 0; k0 < MHA_D; k0 += 32) {
        // --- async stage A tile 16x32 bf16 as 256 b32 words (1 / thread) ---
        {
            int row = tid >> 4, cp = (tid & 15) * 2;
            async_b32(Z + (size_t)(m0 + row) * MHA_D + k0 + cp,
                      &lA[row * 32 + cp]);
        }
        // --- async stage B tile 32x128 f32 transposed ----------------------
#pragma unroll
        for (int t = 0; t < 16; ++t) {
            int i  = tid + t * 256;
            int nn = i & 127, kk = i >> 7;
            async_b32(W + (size_t)(k0 + kk) * MHA_D + n0 + nn,
                      &lBt[nn * BSTR + kk]);
        }
        async_wait0();
        __syncthreads();

        const v8bf* pa = (const v8bf*)(lA + r * 32);
        v16bf af = cat8(pa[2 * hh], pa[2 * hh + 1]);
        const v8f* pb = (const v8f*)(lBt + (w * 16 + r) * BSTR);
        v16bf bfr = cvtf32x16(pb[2 * hh], pb[2 * hh + 1]);

        acc = wmma_bf16(af, bfr, acc);
        __syncthreads();
    }

#pragma unroll
    for (int t = 0; t < 8; ++t) {
        int row = t + hh * 8;
        int gm  = m0 + row;
        int gn  = n0 + w * 16 + r;
        out[(size_t)gm * MHA_D + gn] = acc[t] + bias[gn];
    }
}

// ---------------------------------------------------------------------------
// Host launcher
// ---------------------------------------------------------------------------
extern "C" void kernel_launch(void* const* d_in, const int* in_sizes, int n_in,
                              void* d_out, int out_size, void* d_ws, size_t ws_size,
                              hipStream_t stream) {
    (void)in_sizes; (void)n_in; (void)out_size; (void)ws_size;

    const float* Q    = (const float*)d_in[0];
    const float* K    = (const float*)d_in[1];
    const float* V    = (const float*)d_in[2];
    const float* mask = (const float*)d_in[3];
    const float* wqk  = (const float*)d_in[4];
    const float* wqb  = (const float*)d_in[5];
    const float* outk = (const float*)d_in[6];
    const float* outb = (const float*)d_in[7];

    const size_t NPROJ = (size_t)MHA_B * MHA_H * MHA_S * MHA_HD;  // 4194304
    bf16_t* qbuf  = (bf16_t*)d_ws;
    bf16_t* kbuf  = qbuf + NPROJ;
    bf16_t* vtbuf = kbuf + NPROJ;
    bf16_t* zbuf  = vtbuf + NPROJ;

    float* out_p  = (float*)d_out;                                 // (B,S,D)
    float* attn_p = out_p + (size_t)MHA_B * MHA_S * MHA_D;         // (B,H,S,S)

    dim3 gproj((MHA_B * MHA_S) / 16, MHA_D / 128);
    proj_kernel<<<gproj, 256, 0, stream>>>(Q, wqk, wqb, qbuf, 0);
    proj_kernel<<<gproj, 256, 0, stream>>>(K, wqk, wqb, kbuf, 0);
    proj_kernel<<<gproj, 256, 0, stream>>>(V, wqk, wqb, vtbuf, 1);

    size_t smem = (size_t)(16 * PSTR + 256 + 8 * 8 * 32) * sizeof(float);
    attn_kernel<<<dim3(MHA_S / 16, MHA_H, MHA_B), 256, smem, stream>>>(
        qbuf, kbuf, vtbuf, mask, attn_p, zbuf);

    outproj_kernel<<<gproj, 256, 0, stream>>>(zbuf, outk, outb, out_p);
}
